// SwitchRouter_18184891532042
// MI455X (gfx1250) — matrix operand order repaired and checked by reading, not compiled, not run on hardware
//
#include <hip/hip_runtime.h>
#include <stdint.h>

// ---------------------------------------------------------------------------
// Switch router for MI455X (gfx1250): bf16 WMMA gating GEMM + softmax/argmax,
// TDM-staged w_gate slices in LDS, closed-form capacity enforcement.
// ---------------------------------------------------------------------------

typedef __attribute__((ext_vector_type(16))) __bf16       v16bf;
typedef __attribute__((ext_vector_type(8)))  float        v8f;
typedef __attribute__((ext_vector_type(4)))  unsigned int u32x4;
typedef __attribute__((ext_vector_type(8)))  int          i32x8;
typedef __attribute__((ext_vector_type(4)))  int          i32x4;

#define DGATE 2048
#define EGATE 64
#define KIT   (DGATE / 32)   // 64 K-iterations of 32

// --- Tensor Data Mover: load a 64-row x 32-col bf16 tile of w_gate into LDS.
// D# group0/group1 built per CDNA5 ISA §8.3/8.4 (2D tensor, groups 2/3 zero).
// This toolchain uses the 6-arg builtin form:
//   (uint32x4 g0, int32x8 g1, int32x4, int32x4, int32x8, i32 cpol)
__device__ __forceinline__ void tdm_load_slice(uint32_t lds_off, const void* gptr) {
  uint64_t ga = (uint64_t)(uintptr_t)gptr;
  u32x4 g0 = {
    1u,                                                // count=1, user mode
    lds_off,                                           // lds_addr (bytes)
    (uint32_t)ga,                                      // global_addr[31:0]
    (uint32_t)((ga >> 32) & 0x1FFFFFFull) | (2u << 30) // addr[56:32] | type=2
  };
  i32x8 g1 = {
    (int)(1u << 16),      // workgroup_mask=0, data_size=1 (2 bytes)
    (int)(2048u << 16),   // tensor_dim0 = 2048 (bits 63:48)
    (int)(64u << 16),     // tensor_dim0 hi=0 | tensor_dim1 = 64 (bits 95:80)
    (int)(32u << 16),     // tensor_dim1 hi=0 | tile_dim0 = 32 (bits 127:112)
    64,                   // tile_dim1 = 64, tile_dim2 = 0
    2048,                 // tensor_dim0_stride (elements)
    0, 0                  // dim0_stride hi, tensor_dim1_stride
  };
  i32x4 z4 = {0, 0, 0, 0};
  i32x8 z8 = {0, 0, 0, 0, 0, 0, 0, 0};
  __builtin_amdgcn_tensor_load_to_lds(g0, g1, z4, z4, z8, 0);
}

// --- init: convert w_gate fp32 -> bf16 in ws; zero hist + probsum each call.
__global__ void router_init_kernel(const float* __restrict__ wg,
                                   __bf16* __restrict__ wbf,
                                   float* __restrict__ probsum,
                                   float* __restrict__ hist,
                                   int total, int E) {
  int i = blockIdx.x * blockDim.x + threadIdx.x;
  if (i < total) wbf[i] = (__bf16)wg[i];
  if (i < E) { probsum[i] = 0.f; hist[i] = 0.f; }
}

// --- gating GEMM + softmax + argmax. One wave = 16 tokens x 64 experts.
__global__ __launch_bounds__(256) void router_gate_kernel(
    const float*  __restrict__ x,
    const __bf16* __restrict__ wbf,
    float* __restrict__ w_raw,
    float* __restrict__ idx_out,
    float* __restrict__ hist,
    float* __restrict__ probsum) {
  __shared__ alignas(32) __bf16 wbuf[2][EGATE * 32];  // double-buffered 4 KB slices

  const int tid  = threadIdx.x;
  const int lane = tid & 31;
  const int wave = tid >> 5;
  const int m    = lane & 15;
  const int hi   = lane >> 4;
  const int tile = blockIdx.x * 8 + wave;
  const int row0 = tile * 16;

  // Prologue: wave 0 issues TDM load of slice 0 into slot 0.
  if (tid < 32) {
    tdm_load_slice((uint32_t)(uintptr_t)&wbuf[0][0], wbf);
  }

  const v8f vzero = {0.f, 0.f, 0.f, 0.f, 0.f, 0.f, 0.f, 0.f};
  v8f acc[4];
#pragma unroll
  for (int t = 0; t < 4; ++t) acc[t] = vzero;

  const float* xrow = x + (size_t)(row0 + m) * DGATE;

  for (int it = 0; it < KIT; ++it) {
    if (tid < 32) {
      if (it + 1 < KIT) {
        tdm_load_slice((uint32_t)(uintptr_t)&wbuf[(it + 1) & 1][0],
                       wbf + (size_t)(it + 1) * 32);
        __builtin_amdgcn_s_wait_tensorcnt(1);  // slice `it` has landed
      } else {
        __builtin_amdgcn_s_wait_tensorcnt(0);
      }
    }
    __syncthreads();

    const int k0 = it * 32;
    // A fragment (16x32 bf16): lane holds K = k0 + hi*16 + [0,16).
    const float4* ap  = (const float4*)(xrow + k0 + hi * 16);
    float4 a0 = ap[0], a1 = ap[1];
    const float4* ap2 = (const float4*)(xrow + k0 + 8 + hi * 16);
    float4 a2 = ap2[0], a3 = ap2[1];
    v16bf A;
    A[0]  = (__bf16)a0.x; A[1]  = (__bf16)a0.y; A[2]  = (__bf16)a0.z; A[3]  = (__bf16)a0.w;
    A[4]  = (__bf16)a1.x; A[5]  = (__bf16)a1.y; A[6]  = (__bf16)a1.z; A[7]  = (__bf16)a1.w;
    A[8]  = (__bf16)a2.x; A[9]  = (__bf16)a2.y; A[10] = (__bf16)a2.z; A[11] = (__bf16)a2.w;
    A[12] = (__bf16)a3.x; A[13] = (__bf16)a3.y; A[14] = (__bf16)a3.z; A[15] = (__bf16)a3.w;

    const __bf16* sb = &wbuf[it & 1][0];
#pragma unroll
    for (int t = 0; t < 4; ++t) {
      // B fragment (32x16 bf16): lane holds col n=t*16+m, K = hi*16 + [0,16).
      const v16bf* bp = (const v16bf*)(sb + ((t * 16 + m) * 32 + hi * 16));
      v16bf B = *bp;
      acc[t] = __builtin_amdgcn_wmma_f32_16x16x32_bf16(
          false, A, false, B, (short)0, acc[t], false, false);
    }
    __syncthreads();
  }

  // acc[t][v] = logits[row0 + v + hi*8][t*16 + m]. Reduce over 16-lane halves.
  float rmax[8], rinv[8];
  int   rarg[8];
#pragma unroll
  for (int v = 0; v < 8; ++v) {
    float bv = acc[0][v];
    int   bi = m;
#pragma unroll
    for (int t = 1; t < 4; ++t) {
      float cv = acc[t][v];
      if (cv > bv) { bv = cv; bi = t * 16 + m; }
    }
#pragma unroll
    for (int s = 1; s <= 8; s <<= 1) {
      float ov = __shfl_xor(bv, s, 32);
      int   oi = __shfl_xor(bi, s, 32);
      if (ov > bv || (ov == bv && oi < bi)) { bv = ov; bi = oi; }
    }
    float se = 0.f;
#pragma unroll
    for (int t = 0; t < 4; ++t) se += __expf(acc[t][v] - bv);
#pragma unroll
    for (int s = 1; s <= 8; s <<= 1) se += __shfl_xor(se, s, 32);
    rmax[v] = bv; rarg[v] = bi; rinv[v] = 1.f / se;
  }

  // Per-expert prob sums for aux loss (sum over this tile's 16 tokens).
#pragma unroll
  for (int t = 0; t < 4; ++t) {
    float part = 0.f;
#pragma unroll
    for (int v = 0; v < 8; ++v) part += __expf(acc[t][v] - rmax[v]) * rinv[v];
    part += __shfl_xor(part, 16, 32);
    if (hi == 0) atomicAdd(&probsum[t * 16 + m], part);
  }

  // Lanes 0 and 16 own rows [row0 + hi*8 + v].
  if (m == 0) {
#pragma unroll
    for (int v = 0; v < 8; ++v) {
      int row = row0 + hi * 8 + v;
      w_raw[row]   = rinv[v];            // max prob = 1/sumexp
      idx_out[row] = (float)rarg[v];
      atomicAdd(&hist[rarg[v]], 1.0f);
    }
  }
}

// --- meta: count overflowing experts, last one, and aux loss.
__global__ void router_meta_kernel(const float* __restrict__ hist,
                                   const float* __restrict__ probsum,
                                   float* __restrict__ aux_out,
                                   int* __restrict__ meta,
                                   int E, int N, int cap) {
  if (threadIdx.x == 0 && blockIdx.x == 0) {
    int n_over = 0, e_last = -1;
    float aux = 0.f;
    float capf = (float)cap;
    for (int e = 0; e < E; ++e) {
      float c = hist[e];
      if (c > capf) { n_over++; e_last = e; }
      aux += c * probsum[e];
    }
    aux_out[0] = aux * (float)E / ((float)N * (float)N);
    meta[0] = n_over; meta[1] = e_last;
  }
}

// --- capacity enforcement: closed form of the reference's sequential scan.
//  0 overflows -> unchanged; >1 -> all zero; ==1 -> only top-`cap` tokens
//  (weight desc, index asc) of that expert keep their weight.
__global__ __launch_bounds__(256) void router_enforce_kernel(
    const float* __restrict__ w_raw,
    const float* __restrict__ idxf,
    const int*   __restrict__ meta,
    float* __restrict__ w_out,
    int N, int cap) {
  int i = blockIdx.x * blockDim.x + threadIdx.x;
  if (i >= N) return;
  int n_over = meta[0];
  int e_last = meta[1];
  float w = w_raw[i];
  float out;
  if (n_over == 0) {
    out = w;
  } else if (n_over > 1) {
    out = 0.f;
  } else if ((int)idxf[i] != e_last) {
    out = 0.f;
  } else {
    int rank = 0;
    for (int j = 0; j < N; ++j) {
      if ((int)idxf[j] == e_last) {
        float wj = w_raw[j];
        if (wj > w || (wj == w && j < i)) rank++;
      }
    }
    out = (rank < cap) ? w : 0.f;
  }
  w_out[i] = out;
}

extern "C" void kernel_launch(void* const* d_in, const int* in_sizes, int n_in,
                              void* d_out, int out_size, void* d_ws, size_t ws_size,
                              hipStream_t stream) {
  const float* x  = (const float*)d_in[0];
  const float* wg = (const float*)d_in[1];

  const int D = DGATE;
  const int E = in_sizes[1] / D;        // 64
  const int N = in_sizes[0] / D;        // 16384
  const int cap = (int)((double)N * 1.25 / (double)E);  // 320

  float* out     = (float*)d_out;
  float* w_out   = out;                 // [0, N)
  float* idx_out = out + N;             // [N, 2N)
  float* hist    = out + 2 * N;         // [2N, 2N+E)
  float* aux     = out + 2 * N + E;     // [2N+E]

  char*  wsb     = (char*)d_ws;
  float* w_raw   = (float*)wsb;                          // N floats
  float* probsum = (float*)(wsb + (size_t)N * 4);        // E floats
  int*   meta    = (int*)(wsb + (size_t)N * 4 + (size_t)E * 4);  // 2 ints
  size_t wbf_off = (((size_t)N * 4 + (size_t)E * 4 + 8) + 255) & ~(size_t)255;
  __bf16* wbf    = (__bf16*)(wsb + wbf_off);             // E*D bf16

  const int ED = E * D;
  router_init_kernel<<<(ED + 255) / 256, 256, 0, stream>>>(wg, wbf, probsum, hist, ED, E);
  router_gate_kernel<<<N / 128, 256, 0, stream>>>(x, wbf, w_raw, idx_out, hist, probsum);
  router_meta_kernel<<<1, 32, 0, stream>>>(hist, probsum, aux, meta, E, N, cap);
  router_enforce_kernel<<<(N + 255) / 256, 256, 0, stream>>>(w_raw, idx_out, meta, w_out, N, cap);
}
